// HybridAttentionBlock_11192684773442
// MI455X (gfx1250) — matrix operand order repaired and checked by reading, not compiled, not run on hardware
//
#include <hip/hip_runtime.h>
#include <hip/hip_bf16.h>

// ---------------------------------------------------------------------------
// Problem constants (from reference)
// ---------------------------------------------------------------------------
#define S_LEN 2048
#define DIM   3072
#define NH    12
#define NKV   3
#define NG    (NH / NKV)          // 4
#define LAT   768                 // DIM/4
#define HD    64                  // LAT/NH
#define KVD   192                 // NKV*HD
#define KS    3
#define EPS_  1e-6f
#define GAIN_ 0.25f
#define LOG_THETA 9.210340371976184f   // ln(10000)

// ---------------------------------------------------------------------------
// WMMA / TDM types (CDNA5 gfx1250, wave32)
// ---------------------------------------------------------------------------
typedef __bf16 bf16_t;
typedef bf16_t v16bf __attribute__((ext_vector_type(16)));
typedef float  v8f   __attribute__((ext_vector_type(8)));
typedef unsigned int u32x4 __attribute__((ext_vector_type(4)));
typedef int          i32x4 __attribute__((ext_vector_type(4)));
typedef int          i32x8 __attribute__((ext_vector_type(8)));

#if __has_builtin(__builtin_amdgcn_tensor_load_to_lds)
#define HAVE_TDM 1
#endif

union FragBF16 {
    v16bf v;
    unsigned int u[8];
};

__device__ inline unsigned short f32_to_bf16(float f) {
    union { float f; unsigned int u; } c;
    c.f = f;
    unsigned int u = c.u;
    u += 0x7FFFu + ((u >> 16) & 1u);     // round-to-nearest-even
    return (unsigned short)(u >> 16);
}

__device__ inline unsigned int pack_bf16x2(float a, float b) {
#if __has_builtin(__builtin_amdgcn_cvt_pk_bf16_f32)
    // single packed-convert instruction; type-agnostic bit extraction
    auto t = __builtin_amdgcn_cvt_pk_bf16_f32(a, b);
    union { decltype(t) t2; unsigned int u; } c;
    c.t2 = t;
    return c.u;
#else
    return (unsigned int)f32_to_bf16(a) | ((unsigned int)f32_to_bf16(b) << 16);
#endif
}

// Block-wide sum reduction into `res`; `red` is a 256-entry shared buffer.
#define BLOCK_REDUCE_SUM(val, res)                                   \
    do {                                                             \
        red[tid] = (val);                                            \
        __syncthreads();                                             \
        for (int off_ = 128; off_ > 0; off_ >>= 1) {                 \
            if (tid < off_) red[tid] += red[tid + off_];             \
            __syncthreads();                                         \
        }                                                            \
        (res) = red[0];                                              \
        __syncthreads();                                             \
    } while (0)

// ---------------------------------------------------------------------------
// Tiled bf16 WMMA GEMM:  C[M,N] = A[M,K] * W[N,K]^T   (f32 in memory)
// Block tile 128x64, 256 threads = 8 waves (4 M-waves x 2 N-waves).
// Each wave owns a 32x32 patch: 2 A-frags x 2 B-frags -> 4 WMMAs / K-step.
// Global loads are b128 (float4, load-all-then-convert so loads stay in
// flight); LDS stores are b64 (4 packed bf16).
// Requires M%128==0, N%64==0, K%32==0 (true for all uses here).
// ---------------------------------------------------------------------------
__global__ __launch_bounds__(256)
void gemm_wmma_bf16(const float* __restrict__ A, const float* __restrict__ W,
                    float* __restrict__ C, int M, int N, int K)
{
    __shared__ unsigned short As[128][36];   // 128 rows x 32 K, +4 pad (72B stride)
    __shared__ unsigned short Ws[64][36];

    const int tid  = threadIdx.x;
    const int lane = tid & 31;
    const int wid  = tid >> 5;
    const int half = lane >> 4;
    const int ln   = lane & 15;
    const int wm   = wid & 3;       // 32-row strip (0..3)
    const int wn   = wid >> 2;      // 32-col strip (0..1)

    const int rowBase = blockIdx.y * 128;
    const int colBase = blockIdx.x * 64;

    const int ar = (tid + 0) >> 3, ac = ((tid + 0) & 7) << 2;   // reused each i

    v8f acc[2][2] = {{{}, {}}, {{}, {}}};

    for (int k0 = 0; k0 < K; k0 += 32) {
        // ---- phase 1: issue all global b128 loads ----
        float4 fa[4], fw[2];
        for (int i = 0; i < 4; ++i) {
            int idx = tid + 256 * i;
            int r = idx >> 3, c4 = (idx & 7) << 2;
            fa[i] = *(const float4*)&A[(size_t)(rowBase + r) * K + k0 + c4];
        }
        for (int i = 0; i < 2; ++i) {
            int idx = tid + 256 * i;
            int r = idx >> 3, c4 = (idx & 7) << 2;
            fw[i] = *(const float4*)&W[(size_t)(colBase + r) * K + k0 + c4];
        }
        if (k0 + 32 < K) {   // hint next K-slab (global_prefetch_b8)
            __builtin_prefetch(&A[(size_t)(rowBase + (tid >> 1)) * K + k0 + 32], 0, 1);
            __builtin_prefetch(&W[(size_t)(colBase + (tid >> 2)) * K + k0 + 32], 0, 1);
        }
        // ---- phase 2: convert + LDS store ----
        for (int i = 0; i < 4; ++i) {
            int idx = tid + 256 * i;
            int r = idx >> 3, c4 = (idx & 7) << 2;
            uint2 p; p.x = pack_bf16x2(fa[i].x, fa[i].y);
            p.y = pack_bf16x2(fa[i].z, fa[i].w);
            *(uint2*)&As[r][c4] = p;
        }
        for (int i = 0; i < 2; ++i) {
            int idx = tid + 256 * i;
            int r = idx >> 3, c4 = (idx & 7) << 2;
            uint2 p; p.x = pack_bf16x2(fw[i].x, fw[i].y);
            p.y = pack_bf16x2(fw[i].z, fw[i].w);
            *(uint2*)&Ws[r][c4] = p;
        }
        (void)ar; (void)ac;
        __syncthreads();

        FragBF16 a0, a1, b0, b1;
        // A fragment: lane -> row m=ln ; reg r -> K = (r<4?0:16) + half*8 + 2*(r&3)
        for (int r = 0; r < 8; ++r) {
            int k = ((r < 4) ? 0 : 16) + half * 8 + ((r & 3) << 1);
            a0.u[r] = *(const unsigned int*)&As[wm * 32 + ln][k];
            a1.u[r] = *(const unsigned int*)&As[wm * 32 + 16 + ln][k];
        }
        // B fragment: lane -> col n=ln ; K = (r<4?0:16) + half*8 + 2*(r&3)
        for (int r = 0; r < 8; ++r) {
            int k = ((r < 4) ? 0 : 16) + half * 8 + ((r & 3) << 1);
            b0.u[r] = *(const unsigned int*)&Ws[wn * 32 + ln][k];
            b1.u[r] = *(const unsigned int*)&Ws[wn * 32 + 16 + ln][k];
        }
        acc[0][0] = __builtin_amdgcn_wmma_f32_16x16x32_bf16(false, a0.v, false, b0.v,
                                                            (short)0, acc[0][0], false, false);
        acc[0][1] = __builtin_amdgcn_wmma_f32_16x16x32_bf16(false, a0.v, false, b1.v,
                                                            (short)0, acc[0][1], false, false);
        acc[1][0] = __builtin_amdgcn_wmma_f32_16x16x32_bf16(false, a1.v, false, b0.v,
                                                            (short)0, acc[1][0], false, false);
        acc[1][1] = __builtin_amdgcn_wmma_f32_16x16x32_bf16(false, a1.v, false, b1.v,
                                                            (short)0, acc[1][1], false, false);
        __syncthreads();
    }

    // C layout: reg r -> M = r + 8*half ; lane -> N = ln
    for (int mi = 0; mi < 2; ++mi)
        for (int ni = 0; ni < 2; ++ni)
            for (int r = 0; r < 8; ++r) {
                int row = rowBase + wm * 32 + mi * 16 + r + 8 * half;
                int col = colBase + wn * 32 + ni * 16 + ln;
                C[(size_t)row * N + col] = acc[mi][ni][r];
            }
}

// ---------------------------------------------------------------------------
// Row-wise rmsnorm: out[row,:] = in[row,:] * rsqrt(mean(in^2)+eps) * w
// ---------------------------------------------------------------------------
__global__ __launch_bounds__(256)
void rmsnorm_rows(const float* __restrict__ in, const float* __restrict__ w,
                  float* __restrict__ out, int width)
{
    const int row = blockIdx.x;
    const int tid = threadIdx.x;
    __shared__ float red[256];

    float ss = 0.f;
    for (int j = tid; j < width; j += 256) {
        float v = in[(size_t)row * width + j];
        ss += v * v;
    }
    float tot;
    BLOCK_REDUCE_SUM(ss, tot);
    float r = rsqrtf(tot / (float)width + EPS_);
    for (int j = tid; j < width; j += 256)
        out[(size_t)row * width + j] = in[(size_t)row * width + j] * r * w[j];
}

// ---------------------------------------------------------------------------
// Grouped causal conv, kernel size 3, 64-channel groups.
// ---------------------------------------------------------------------------
__global__ __launch_bounds__(256)
void conv_causal(const float* __restrict__ xin, const float* __restrict__ w,
                 float* __restrict__ out, int width /*LAT or KVD*/)
{
    const int row = blockIdx.x;            // b*S + s
    const int s   = row % S_LEN;
    const int tid = threadIdx.x;
    __shared__ float xr[3][LAT];           // max width

    for (int t = 0; t < 3; ++t) {
        int ss = s - 2 + t;
        for (int j = tid; j < width; j += 256)
            xr[t][j] = (ss >= 0) ? xin[(size_t)(row - 2 + t) * width + j] : 0.f;
    }
    __syncthreads();

    for (int oc = tid; oc < width; oc += 256) {
        int g = oc >> 6;
        const float* wp = w + (size_t)oc * 64 * KS;
        float acc = 0.f;
        for (int cin = 0; cin < 64; ++cin) {
            int c = (g << 6) + cin;
            acc += wp[cin * 3 + 0] * xr[0][c]
                 + wp[cin * 3 + 1] * xr[1][c]
                 + wp[cin * 3 + 2] * xr[2][c];
        }
        out[(size_t)row * width + oc] = acc;
    }
}

// ---------------------------------------------------------------------------
// Fused per-token prep: conv-rmsnorm -> head-mean mixing -> pm rmsnorm ->
// per-head l2norm -> RoPE -> bf16 pack into [B,H,S,HD] for attention.
// Attention scale HD^-0.5 * sqrt(HD) == 1 is folded into q; k keeps 8*key_temp.
// ---------------------------------------------------------------------------
__global__ __launch_bounds__(256)
void prep_kernel(const float* __restrict__ qp,  // q_pre  (BS,768) normalized
                 const float* __restrict__ kp,  // k_pre  (BS,192) normalized
                 const float* __restrict__ vp,  // v      (BS,192) normalized
                 const float* __restrict__ qc,  // conv q (BS,768) raw
                 const float* __restrict__ kc,  // conv k (BS,192) raw
                 const float* __restrict__ convn_w, const float* __restrict__ kconvn_w,
                 const float* __restrict__ pmq_w,   const float* __restrict__ pmk_w,
                 const float* __restrict__ key_temp,
                 unsigned short* __restrict__ qb,   // [B,NH,S,HD] bf16
                 unsigned short* __restrict__ kb,   // [B,NKV,S,HD] bf16
                 unsigned short* __restrict__ vb)   // [B,NKV,S,HD] bf16
{
    const int row = blockIdx.x;
    const int b = row / S_LEN, s = row % S_LEN;
    const int tid = threadIdx.x;
    __shared__ float red[256];
    __shared__ float qmean[64], kmean[64];
    __shared__ float qrow[LAT], krow[KVD];
    __shared__ float hsum[16];

    if (tid < 64) {
        float aq = 0.f, ak = 0.f;
        for (int hh = 0; hh < NH;  ++hh) aq += qp[(size_t)row * LAT + hh * 64 + tid];
        for (int hh = 0; hh < NKV; ++hh) ak += kp[(size_t)row * KVD + hh * 64 + tid];
        qmean[tid] = aq * (1.f / NH);
        kmean[tid] = ak * (1.f / NKV);
    }

    float v0_ = qc[(size_t)row * LAT + tid];
    float v1_ = qc[(size_t)row * LAT + tid + 256];
    float v2_ = qc[(size_t)row * LAT + tid + 512];
    float kcv = (tid < KVD) ? kc[(size_t)row * KVD + tid] : 0.f;

    float tot;
    BLOCK_REDUCE_SUM(v0_ * v0_ + v1_ * v1_ + v2_ * v2_, tot);
    float rq = rsqrtf(tot / (float)LAT + EPS_);
    BLOCK_REDUCE_SUM(kcv * kcv, tot);
    float rk = rsqrtf(tot / (float)KVD + EPS_);

    // conv-norm + mean mixing (qmean/kmean visible: barriers inside reduces)
    float km = kmean[tid & 63];
    float q2a = v0_ * rq * convn_w[tid]       + GAIN_ * km;
    float q2b = v1_ * rq * convn_w[tid + 256] + GAIN_ * km;
    float q2c = v2_ * rq * convn_w[tid + 512] + GAIN_ * km;
    float k2  = 0.f;
    if (tid < KVD) k2 = kcv * rk * kconvn_w[tid] + GAIN_ * qmean[tid & 63];

    BLOCK_REDUCE_SUM(q2a * q2a + q2b * q2b + q2c * q2c, tot);
    float rq2 = rsqrtf(tot / (float)LAT + EPS_);
    BLOCK_REDUCE_SUM(k2 * k2, tot);
    float rk2 = rsqrtf(tot / (float)KVD + EPS_);

    qrow[tid]       = q2a * rq2 * pmq_w[tid];
    qrow[tid + 256] = q2b * rq2 * pmq_w[tid + 256];
    qrow[tid + 512] = q2c * rq2 * pmq_w[tid + 512];
    if (tid < KVD) krow[tid] = k2 * rk2 * pmk_w[tid];
    __syncthreads();

    if (tid < NH) {                       // per-head l2 sums for q
        float hs = 0.f;
        for (int d = 0; d < HD; ++d) { float t = qrow[tid * HD + d]; hs += t * t; }
        hsum[tid] = hs;
    } else if (tid < NH + NKV) {          // for k
        int hh = tid - NH;
        float hs = 0.f;
        for (int d = 0; d < HD; ++d) { float t = krow[hh * HD + d]; hs += t * t; }
        hsum[tid] = hs;
    }
    __syncthreads();

    const float ktemp = key_temp[0];
    const float fs = (float)s;

    for (int j = tid; j < LAT; j += 256) {    // q: scale folded to 1.0
        int hh = j >> 6, d = j & 63, de = d & ~1;
        float inv = 1.f / fmaxf(sqrtf(hsum[hh]), 1e-12f);
        float x1 = qrow[(hh << 6) + de]     * inv;
        float x2 = qrow[(hh << 6) + de + 1] * inv;
        float fr  = __expf(-((float)de / (float)HD) * LOG_THETA);
        float ang = fs * fr;
        float cs = __cosf(ang), sn = __sinf(ang);
        float res = (d & 1) ? (x1 * sn + x2 * cs) : (x1 * cs - x2 * sn);
        qb[((size_t)(b * NH + hh) * S_LEN + s) * HD + d] = f32_to_bf16(res);
    }
    if (tid < KVD) {                           // k: 8*key_temp ; v: passthrough
        int hh = tid >> 6, d = tid & 63, de = d & ~1;
        float inv = 8.f * ktemp / fmaxf(sqrtf(hsum[NH + hh]), 1e-12f);
        float x1 = krow[(hh << 6) + de]     * inv;
        float x2 = krow[(hh << 6) + de + 1] * inv;
        float fr  = __expf(-((float)de / (float)HD) * LOG_THETA);
        float ang = fs * fr;
        float cs = __cosf(ang), sn = __sinf(ang);
        float res = (d & 1) ? (x1 * sn + x2 * cs) : (x1 * cs - x2 * sn);
        kb[((size_t)(b * NKV + hh) * S_LEN + s) * HD + d] = f32_to_bf16(res);
        vb[((size_t)(b * NKV + hh) * S_LEN + s) * HD + d] =
            f32_to_bf16(vp[(size_t)row * KVD + tid]);
    }
}

// ---------------------------------------------------------------------------
// Flash attention (causal, GQA NH=12 over NKV=3, HD=64).
// grid = (S/64, NH, B), 128 threads = 4 waves; wave owns 16 query rows.
// K tile is fetched by the Tensor Data Mover (tensor_load_to_lds) with LDS
// row padding 128B->144B; V tile is loaded b128 + transposed into LDS.
// QK^T and P*V via v_wmma_f32_16x16x32_bf16; online softmax via shfl_xor
// across the 16-lane column groups of the C fragment (wave32).
// ---------------------------------------------------------------------------
__global__ __launch_bounds__(128)
void flash_attn_kernel(const unsigned short* __restrict__ qb,
                       const unsigned short* __restrict__ kb,
                       const unsigned short* __restrict__ vb,
                       float* __restrict__ out /* [B,S,NH,HD] */)
{
    const int qt  = blockIdx.x;
    const int h   = blockIdx.y;
    const int b   = blockIdx.z;
    const int kvh = h / NG;
    const int tid = threadIdx.x;
    const int wid = tid >> 5, lane = tid & 31;
    const int half = lane >> 4, ln = lane & 15;
    const int qbase = qt * 64;

    __shared__ unsigned short Kt[64][72];       // [key][hd]   (144B row stride)
    __shared__ unsigned short Vt[64][72];       // [hd][key]   (transposed)
    __shared__ unsigned short Ps[4][16][72];    // per-wave P stage [m][key]

    // ---- load Q fragments (A layout), 2 K-chunks of HD ----
    unsigned int qf[2][8];
    {
        const unsigned short* qptr =
            qb + ((size_t)(b * NH + h) * S_LEN + qbase + wid * 16 + ln) * HD;
        for (int kc = 0; kc < 2; ++kc)
            for (int r = 0; r < 8; ++r) {
                int k = kc * 32 + ((r < 4) ? 0 : 16) + half * 8 + ((r & 3) << 1);
                qf[kc][r] = *(const unsigned int*)(qptr + k);
            }
    }

    v8f o[4] = {{}, {}, {}, {}};
    float mrow[8], lrow[8];
    for (int r = 0; r < 8; ++r) { mrow[r] = -3.0e38f; lrow[r] = 0.f; }

    for (int kt = 0; kt <= qt; ++kt) {
        const unsigned short* kptr = kb + ((size_t)(b * NKV + kvh) * S_LEN + kt * 64) * HD;
        const unsigned short* vptr = vb + ((size_t)(b * NKV + kvh) * S_LEN + kt * 64) * HD;

#ifdef HAVE_TDM
        // ---- K tile via Tensor Data Mover (wave 0 issues; TDM ignores EXEC) ----
        if (wid == 0) {
            unsigned long long ga = (unsigned long long)kptr;
            unsigned int lds_k = (unsigned int)(unsigned long long)&Kt[0][0];
            u32x4 g0 = { 1u,                     // count=1, user descriptor
                         lds_k,                  // lds_addr
                         (unsigned int)ga,       // global_addr[31:0]
                         ((unsigned int)(ga >> 32) & 0x1FFFFFFu) | 0x80000000u }; // +type=2
            i32x8 g1 = { (int)((1u << 16)        // data_size = 2 bytes
                             | (1u << 20)        // pad_enable
                             | (4u << 22)        // pad_interval = 32 dwords (one row)
                             | (3u << 25)),      // pad_amount   = 4 dwords (16B)
                         (int)(64u << 16),       // tensor_dim0 = 64
                         (int)(64u << 16),       // tensor_dim1 = 64
                         (int)(64u << 16),       // tile_dim0   = 64
                         64,                     // tile_dim1   = 64
                         64,                     // tensor_dim0_stride = 64
                         0, 0 };
            i32x4 z4 = {0, 0, 0, 0};
#if defined(__clang_major__) && (__clang_major__ >= 23)
            i32x8 z8 = {0, 0, 0, 0, 0, 0, 0, 0};
            __builtin_amdgcn_tensor_load_to_lds(g0, g1, z4, z4, z8, 0);
#else
            __builtin_amdgcn_tensor_load_to_lds(g0, g1, z4, z4, 0);
#endif
        }
#else
        // fallback: cooperative K tile copy
        for (int i = 0; i < 4; ++i) {
            int lin = tid + 128 * i;                 // 512 x 8 shorts
            int key = lin >> 3, c8 = (lin & 7) << 3;
            *(uint4*)&Kt[key][c8] = *(const uint4*)(kptr + key * HD + c8);
        }
#endif
        // ---- V tile: b128 loads (all in flight), then transpose into LDS ----
        {
            uint4 vv[4];
            for (int i = 0; i < 4; ++i) {
                int lin = tid + 128 * i;
                int key = lin >> 3, c8 = (lin & 7) << 3;
                vv[i] = *(const uint4*)(vptr + key * HD + c8);
            }
            for (int i = 0; i < 4; ++i) {
                int lin = tid + 128 * i;
                int key = lin >> 3, c8 = (lin & 7) << 3;
                Vt[c8 + 0][key] = (unsigned short)(vv[i].x & 0xffffu);
                Vt[c8 + 1][key] = (unsigned short)(vv[i].x >> 16);
                Vt[c8 + 2][key] = (unsigned short)(vv[i].y & 0xffffu);
                Vt[c8 + 3][key] = (unsigned short)(vv[i].y >> 16);
                Vt[c8 + 4][key] = (unsigned short)(vv[i].z & 0xffffu);
                Vt[c8 + 5][key] = (unsigned short)(vv[i].z >> 16);
                Vt[c8 + 6][key] = (unsigned short)(vv[i].w & 0xffffu);
                Vt[c8 + 7][key] = (unsigned short)(vv[i].w >> 16);
            }
        }
#ifdef HAVE_TDM
        if (wid == 0) {
#if __has_builtin(__builtin_amdgcn_s_wait_tensorcnt)
            __builtin_amdgcn_s_wait_tensorcnt(0);
#else
            asm volatile("s_wait_tensorcnt 0x0" ::: "memory");
#endif
        }
#endif
        __syncthreads();

        // ---- scores: 16x64 per wave ----
        v8f sc[4];
        for (int n = 0; n < 4; ++n) {
            v8f acc = {};
            for (int kc = 0; kc < 2; ++kc) {
                FragBF16 a, bf_;
                for (int r = 0; r < 8; ++r) a.u[r] = qf[kc][r];
                int key = n * 16 + ln;
                for (int r = 0; r < 8; ++r) {
                    int hd = kc * 32 + ((r < 4) ? 0 : 16) + half * 8 + ((r & 3) << 1);
                    bf_.u[r] = *(const unsigned int*)&Kt[key][hd];
                }
                acc = __builtin_amdgcn_wmma_f32_16x16x32_bf16(
                          false, a.v, false, bf_.v, (short)0, acc, false, false);
            }
            sc[n] = acc;
        }

        // ---- causal mask on the diagonal tile ----
        if (kt == qt) {
            for (int n = 0; n < 4; ++n)
                for (int r = 0; r < 8; ++r) {
                    int qi = wid * 16 + r + 8 * half;   // local: kbase==qbase
                    int kj = n * 16 + ln;
                    if (kj > qi) sc[n][r] = -3.0e38f;
                }
        }

        // ---- online softmax ----
        for (int r = 0; r < 8; ++r) {
            float mx = fmaxf(fmaxf(sc[0][r], sc[1][r]), fmaxf(sc[2][r], sc[3][r]));
            for (int m = 1; m < 16; m <<= 1) mx = fmaxf(mx, __shfl_xor(mx, m, 32));
            float nm   = fmaxf(mrow[r], mx);
            float corr = __expf(mrow[r] - nm);
            float rsum = 0.f;
            for (int n = 0; n < 4; ++n) {
                float p = __expf(sc[n][r] - nm);
                sc[n][r] = p;
                rsum += p;
            }
            for (int m = 1; m < 16; m <<= 1) rsum += __shfl_xor(rsum, m, 32);
            lrow[r] = lrow[r] * corr + rsum;
            mrow[r] = nm;
            for (int n = 0; n < 4; ++n) o[n][r] *= corr;
        }

        // ---- stage P (C layout -> LDS -> A layout) ----
        for (int n = 0; n < 4; ++n)
            for (int r = 0; r < 8; ++r)
                Ps[wid][r + 8 * half][n * 16 + ln] = f32_to_bf16(sc[n][r]);
        __syncthreads();

        // ---- O += P * V ----
        for (int n = 0; n < 4; ++n) {
            for (int kc = 0; kc < 2; ++kc) {
                FragBF16 a, bf_;
                for (int r = 0; r < 8; ++r) {
                    int k = kc * 32 + ((r < 4) ? 0 : 16) + half * 8 + ((r & 3) << 1);
                    a.u[r] = *(const unsigned int*)&Ps[wid][ln][k];
                }
                for (int r = 0; r < 8; ++r) {
                    int key = kc * 32 + ((r < 4) ? 0 : 16) + half * 8 + ((r & 3) << 1);
                    bf_.u[r] = *(const unsigned int*)&Vt[n * 16 + ln][key];
                }
                o[n] = __builtin_amdgcn_wmma_f32_16x16x32_bf16(
                           false, a.v, false, bf_.v, (short)0, o[n], false, false);
            }
        }
        __syncthreads();
    }

    // ---- epilogue: normalize and write [b, s, h, d] ----
    for (int n = 0; n < 4; ++n)
        for (int r = 0; r < 8; ++r) {
            int m  = r + 8 * half;
            int qi = qbase + wid * 16 + m;
            float val = o[n][r] / lrow[r];
            out[(((size_t)b * S_LEN + qi) * NH + h) * HD + n * 16 + ln] = val;
        }
}

// ---------------------------------------------------------------------------
// Host-side orchestration
// ---------------------------------------------------------------------------
extern "C" void kernel_launch(void* const* d_in, const int* in_sizes, int n_in,
                              void* d_out, int out_size, void* d_ws, size_t ws_size,
                              hipStream_t stream) {
    const float* x        = (const float*)d_in[0];
    const float* wq       = (const float*)d_in[1];
    const float* wk       = (const float*)d_in[2];
    const float* wv       = (const float*)d_in[3];
    const float* ln_w     = (const float*)d_in[4];
    const float* kvln_w   = (const float*)d_in[5];
    const float* qconv_w  = (const float*)d_in[6];
    const float* kconv_w  = (const float*)d_in[7];
    const float* convn_w  = (const float*)d_in[8];
    const float* kconvn_w = (const float*)d_in[9];
    const float* pmq_w    = (const float*)d_in[10];
    const float* pmk_w    = (const float*)d_in[11];
    const float* key_temp = (const float*)d_in[12];
    const float* preout_w = (const float*)d_in[13];
    const float* wo       = (const float*)d_in[14];
    float* outp           = (float*)d_out;

    const int BS = in_sizes[0] / DIM;      // B * S  (= 4096)
    const int B  = BS / S_LEN;             // (= 2)

    // workspace carve-up (all offsets 16B-aligned)
    float* q0        = (float*)d_ws;                 // BS*LAT
    float* k0        = q0 + (size_t)BS * LAT;        // BS*KVD
    float* v0        = k0 + (size_t)BS * KVD;        // BS*KVD
    float* qc        = v0 + (size_t)BS * KVD;        // BS*LAT
    float* kc        = qc + (size_t)BS * LAT;        // BS*KVD
    float* attn_out  = kc + (size_t)BS * KVD;        // BS*LAT
    float* attn_norm = attn_out + (size_t)BS * LAT;  // BS*LAT
    unsigned short* qbv = (unsigned short*)(attn_norm + (size_t)BS * LAT); // BS*LAT
    unsigned short* kbv = qbv + (size_t)BS * LAT;    // BS*KVD
    unsigned short* vbv = kbv + (size_t)BS * KVD;    // BS*KVD

    // 1) projections (WMMA GEMMs)
    gemm_wmma_bf16<<<dim3(LAT / 64, BS / 128), 256, 0, stream>>>(x, wq, q0, BS, LAT, DIM);
    gemm_wmma_bf16<<<dim3(KVD / 64, BS / 128), 256, 0, stream>>>(x, wk, k0, BS, KVD, DIM);
    gemm_wmma_bf16<<<dim3(KVD / 64, BS / 128), 256, 0, stream>>>(x, wv, v0, BS, KVD, DIM);

    // 2) post-projection rmsnorms (in place)
    rmsnorm_rows<<<BS, 256, 0, stream>>>(q0, ln_w,   q0, LAT);
    rmsnorm_rows<<<BS, 256, 0, stream>>>(k0, kvln_w, k0, KVD);
    rmsnorm_rows<<<BS, 256, 0, stream>>>(v0, kvln_w, v0, KVD);

    // 3) grouped causal convs
    conv_causal<<<BS, 256, 0, stream>>>(q0, qconv_w, qc, LAT);
    conv_causal<<<BS, 256, 0, stream>>>(k0, kconv_w, kc, KVD);

    // 4) fused norm/mix/l2/rope/pack
    prep_kernel<<<BS, 256, 0, stream>>>(q0, k0, v0, qc, kc, convn_w, kconvn_w,
                                        pmq_w, pmk_w, key_temp, qbv, kbv, vbv);

    // 5) flash attention (WMMA + TDM K-tile fetch)
    flash_attn_kernel<<<dim3(S_LEN / 64, NH, B), 128, 0, stream>>>(qbv, kbv, vbv, attn_out);

    // 6) pre-output rmsnorm + output projection (WMMA GEMM)
    rmsnorm_rows<<<BS, 256, 0, stream>>>(attn_out, preout_w, attn_norm, LAT);
    gemm_wmma_bf16<<<dim3(DIM / 64, BS / 128), 256, 0, stream>>>(attn_norm, wo, outp,
                                                                 BS, DIM, LAT);
}
